// bilstm_crf_70368744178310
// MI455X (gfx1250) — matrix operand order repaired and checked by reading, not compiled, not run on hardware
//
#include <hip/hip_runtime.h>
#include <hip/hip_bf16.h>
#include <math.h>

// ---- problem constants (from reference) ----
#define VOCAB_ 50000
#define EMB_   50
#define H_     200
#define T_     20
#define B_     64
#define S_     512
#define BS_    (B_ * S_)        // 32768
#define G_     (4 * H_)         // 800  (gate width)
#define H2_    (2 * H_)         // 400
#define K0_    64               // EMB padded to 2 k-steps of 32
#define K1_    416              // 2H padded to 13 k-steps of 32
#define KH_    224              // H padded to 7 k-steps of 32
#define TP_    32               // T padded for WMMA N

typedef __attribute__((ext_vector_type(16))) __bf16 v16bf;
typedef __attribute__((ext_vector_type(8)))  float  v8f;
typedef __attribute__((ext_vector_type(8)))  __bf16 bf16x8;

// ---------------- fragment loaders (wave32 WMMA bf16 layouts) ----------------
// A (16x32, bf16 row-major): lane r/half: elems 0..7 -> K=kk+half*8+e, 8..15 -> K=kk+16+half*8+e
__device__ __forceinline__ v16bf load_a_frag(const __bf16* __restrict__ A, int Kp,
                                             int row, int kk, int half) {
  const bf16x8* p0 = (const bf16x8*)(A + (size_t)row * Kp + kk + half * 8);
  const bf16x8* p1 = (const bf16x8*)(A + (size_t)row * Kp + kk + 16 + half * 8);
  bf16x8 g0 = *p0, g1 = *p1;
  v16bf a;
#pragma unroll
  for (int e = 0; e < 8; ++e) { a[e] = g0[e]; a[8 + e] = g1[e]; }
  return a;
}
// B packed fragment-major: for k-tile kt and column n, the 32 K values are contiguous
// at Bp[(kt*Np + n)*32 + ko]. Lane (half,r) needs K = kt*32 + half*8 + e (e<8) and
// kt*32 + 16 + half*8 + (e-8) (e>=8) -> two 16B vector loads.
__device__ __forceinline__ v16bf load_b_frag_packed(const __bf16* __restrict__ Bp, int Np,
                                                    int col, int kt, int half) {
  const bf16x8* p = (const bf16x8*)(Bp + (((size_t)kt * Np + col) << 5) + half * 8);
  bf16x8 g0 = p[0], g1 = p[2];
  v16bf b;
#pragma unroll
  for (int e = 0; e < 8; ++e) { b[e] = g0[e]; b[8 + e] = g1[e]; }
  return b;
}

__device__ __forceinline__ float sigmf(float x) { return 1.0f / (1.0f + expf(-x)); }

// ---------------- utility kernels ----------------
__global__ void fill_zero_k(float* __restrict__ p, long n) {
  long id = (long)blockIdx.x * blockDim.x + threadIdx.x;
  if (id < n) p[id] = 0.0f;
}

// pack W (Norig x Korig, f32) -> bf16 fragment-major B: out[(kt*Np+n)*32+ko] = W[n][kt*32+ko]
__global__ void pack_wT_k(const float* __restrict__ W, __bf16* __restrict__ out,
                          int Korig, int Kp, int Norig, int Np) {
  int id = blockIdx.x * blockDim.x + threadIdx.x;
  if (id >= Kp * Np) return;
  int k = id / Np, n = id % Np;
  int kt = k >> 5, ko = k & 31;
  float v = (k < Korig && n < Norig) ? W[(size_t)n * Korig + k] : 0.0f;
  out[(((size_t)kt * Np + n) << 5) + ko] = (__bf16)v;
}

__global__ void bias_sum_k(const float* __restrict__ a, const float* __restrict__ b,
                           float* __restrict__ out, int n) {
  int id = blockIdx.x * blockDim.x + threadIdx.x;
  if (id < n) out[id] = a[id] + b[id];
}

// pad bias to Np entries with zeros (uniform epilogue, no divergence)
__global__ void pad_bias_k(const float* __restrict__ b, float* __restrict__ out,
                           int Norig, int Np) {
  int id = blockIdx.x * blockDim.x + threadIdx.x;
  if (id < Np) out[id] = (id < Norig) ? b[id] : 0.0f;
}

// embedding gather -> bf16 X0, time-major rows: row = s*B + b, padded to K0_ cols
__global__ void embed_gather_k(const int* __restrict__ x, const float* __restrict__ emb,
                               __bf16* __restrict__ X0) {
  int id = blockIdx.x * blockDim.x + threadIdx.x;
  if (id >= BS_ * K0_) return;
  int row = id >> 6, k = id & 63;
  int b = row % B_, s = row / B_;
  float v = 0.0f;
  if (k < EMB_) v = emb[(size_t)x[b * S_ + s] * EMB_ + k];
  X0[id] = (__bf16)v;
}

// ---------------- generic bf16 WMMA GEMM (compile-time K): C = A @ B (+biasPad) ---------
// A: M x KP bf16 row-major ; Bp: packed fragment-major ; C: M x Np f32
template <int KP>
__global__ void gemm_bf16_k(const __bf16* __restrict__ A, const __bf16* __restrict__ Bp,
                            const float* __restrict__ biasPad, float* __restrict__ C,
                            int M, int Np) {
  int wave = blockIdx.x * (blockDim.x >> 5) + (threadIdx.x >> 5);
  int lane = threadIdx.x & 31;
  int half = lane >> 4, r = lane & 15;
  int tilesN = Np >> 4;
  int tiles = (M >> 4) * tilesN;
  if (wave >= tiles) return;  // wave-uniform: EXEC stays all-ones for WMMA
  int tm = wave / tilesN, tn = wave % tilesN;
  int row = tm * 16 + r, col = tn * 16 + r;
  v8f acc = {};
#pragma unroll
  for (int kt = 0; kt < KP / 32; ++kt) {
    if (kt + 1 < KP / 32)
      __builtin_prefetch(A + (size_t)row * KP + (kt + 1) * 32, 0, 0);
    v16bf a = load_a_frag(A, KP, row, kt * 32, half);
    v16bf b = load_b_frag_packed(Bp, Np, col, kt, half);
    acc = __builtin_amdgcn_wmma_f32_16x16x32_bf16(false, a, false, b, (short)0, acc,
                                                  false, false);
  }
#pragma unroll
  for (int v = 0; v < 8; ++v) {
    int m = tm * 16 + v + half * 8;
    float o = acc[v];
    if (biasPad) o += biasPad[col];  // uniform branch; biasPad is zero-padded
    C[(size_t)m * Np + col] = o;
  }
}

// ---------------- LSTM recurrent step: gates = h @ Whh^T + pre[t] + bias ----------------
// grid(50, 2): blockIdx.x = N-tile (G_/16), blockIdx.y = dir; block 128 = 4 waves
__global__ void lstm_step_gemm_k(const __bf16* __restrict__ hbf,  // [2][B_*KH_]
                                 const __bf16* __restrict__ WhhP_f,
                                 const __bf16* __restrict__ WhhP_b,
                                 const float* __restrict__ pre_f,
                                 const float* __restrict__ pre_b,
                                 const float* __restrict__ bias_f,
                                 const float* __restrict__ bias_b,
                                 float* __restrict__ gbuf, int t) {
  int dir = blockIdx.y;
  int tn = blockIdx.x;
  int tm = threadIdx.x >> 5;
  int lane = threadIdx.x & 31;
  int half = lane >> 4, r = lane & 15;
  const __bf16* A = hbf + (size_t)dir * (B_ * KH_);
  const __bf16* Bp = dir ? WhhP_b : WhhP_f;
  const float* pre = dir ? pre_b : pre_f;
  const float* bias = dir ? bias_b : bias_f;
  int tt = dir ? (S_ - 1 - t) : t;
  int row = tm * 16 + r, col = tn * 16 + r;
  v8f acc = {};
#pragma unroll
  for (int kt = 0; kt < KH_ / 32; ++kt) {
    v16bf a = load_a_frag(A, KH_, row, kt * 32, half);
    v16bf b = load_b_frag_packed(Bp, G_, col, kt, half);
    acc = __builtin_amdgcn_wmma_f32_16x16x32_bf16(false, a, false, b, (short)0, acc,
                                                  false, false);
  }
  float* gout = gbuf + (size_t)dir * (B_ * G_);
  const float* prerow = pre + (size_t)tt * B_ * G_;
  float bv = bias[col];
#pragma unroll
  for (int v = 0; v < 8; ++v) {
    int b = tm * 16 + v + half * 8;
    gout[b * G_ + col] = acc[v] + prerow[b * G_ + col] + bv;
  }
}

// pointwise gate update; writes bf16 h into per-step state and layer-output X
__global__ void lstm_step_point_k(const float* __restrict__ gbuf, float* __restrict__ cstate,
                                  __bf16* __restrict__ hbf, __bf16* __restrict__ Xout, int t) {
  int id = blockIdx.x * blockDim.x + threadIdx.x;
  if (id >= 2 * B_ * H_) return;
  int dir = id / (B_ * H_);
  int rem = id % (B_ * H_);
  int b = rem / H_, k = rem % H_;
  const float* g = gbuf + (size_t)dir * (B_ * G_) + (size_t)b * G_;
  float gi = g[k], gf = g[H_ + k], gg = g[2 * H_ + k], go = g[3 * H_ + k];
  float c = cstate[id];
  c = sigmf(gf) * c + sigmf(gi) * tanhf(gg);
  float h = sigmf(go) * tanhf(c);
  cstate[id] = c;
  hbf[(size_t)dir * (B_ * KH_) + b * KH_ + k] = (__bf16)h;
  int tt = dir ? (S_ - 1 - t) : t;
  Xout[((size_t)tt * B_ + b) * K1_ + dir * H_ + k] = (__bf16)h;
}

// ---------------- CRF: one wave per batch row ----------------
__global__ void crf_k(const float* __restrict__ em, const int* __restrict__ y,
                      const unsigned char* __restrict__ mask,
                      const float* __restrict__ start, const float* __restrict__ endv,
                      const float* __restrict__ trans, float* __restrict__ loglik) {
  __shared__ float tr[T_ * T_];
  __shared__ float alpha[T_];
  __shared__ float nalpha[T_];
  int b = blockIdx.x;
  int tid = threadIdx.x;
  for (int i = tid; i < T_ * T_; i += 32) tr[i] = trans[i];
  __syncthreads();
  if (tid < T_) alpha[tid] = start[tid] + em[(size_t)b * TP_ + tid];
  __syncthreads();
  for (int t = 1; t < S_; ++t) {
    if (tid < T_) {
      float mx = -1e30f;
#pragma unroll
      for (int i = 0; i < T_; ++i) mx = fmaxf(mx, alpha[i] + tr[i * T_ + tid]);
      float s = 0.0f;
#pragma unroll
      for (int i = 0; i < T_; ++i) s += expf(alpha[i] + tr[i * T_ + tid] - mx);
      float val = mx + logf(s) + em[((size_t)t * B_ + b) * TP_ + tid];
      nalpha[tid] = mask[b * S_ + t] ? val : alpha[tid];
    }
    __syncthreads();
    if (tid < T_) alpha[tid] = nalpha[tid];
    __syncthreads();
  }
  if (tid == 0) {
    float mx = -1e30f;
    for (int j = 0; j < T_; ++j) mx = fmaxf(mx, alpha[j] + endv[j]);
    float s = 0.0f;
    for (int j = 0; j < T_; ++j) s += expf(alpha[j] + endv[j] - mx);
    float denom = mx + logf(s);
    int y0 = y[b * S_];
    float num = start[y0] + em[(size_t)b * TP_ + y0];
    int prev = y0;
    int seqlen = mask[b * S_] ? 1 : 0;
    for (int t = 1; t < S_; ++t) {
      int yt = y[b * S_ + t];
      float m = mask[b * S_ + t] ? 1.0f : 0.0f;
      num += m * (tr[prev * T_ + yt] + em[((size_t)t * B_ + b) * TP_ + yt]);
      prev = yt;
      seqlen += mask[b * S_ + t] ? 1 : 0;
    }
    int last = y[b * S_ + (seqlen > 0 ? seqlen - 1 : 0)];
    num += endv[last];
    loglik[b] = num - denom;
  }
}

// deterministic ordered reduction (no float atomics -> bit-stable across replays)
__global__ void final_sum_k(const float* __restrict__ loglik, float* __restrict__ out) {
  if (blockIdx.x == 0 && threadIdx.x == 0) {
    float s = 0.0f;
    for (int i = 0; i < B_; ++i) s += loglik[i];
    out[0] = s;
  }
}

// ---------------- host launcher ----------------
extern "C" void kernel_launch(void* const* d_in, const int* in_sizes, int n_in,
                              void* d_out, int out_size, void* d_ws, size_t ws_size,
                              hipStream_t stream) {
  const int* x = (const int*)d_in[0];
  const int* y = (const int*)d_in[1];
  const unsigned char* mask = (const unsigned char*)d_in[2];
  const float* emb = (const float*)d_in[3];
  const float* Wih[2][2] = {{(const float*)d_in[4], (const float*)d_in[8]},
                            {(const float*)d_in[12], (const float*)d_in[16]}};
  const float* Whh[2][2] = {{(const float*)d_in[5], (const float*)d_in[9]},
                            {(const float*)d_in[13], (const float*)d_in[17]}};
  const float* bih[2][2] = {{(const float*)d_in[6], (const float*)d_in[10]},
                            {(const float*)d_in[14], (const float*)d_in[18]}};
  const float* bhh[2][2] = {{(const float*)d_in[7], (const float*)d_in[11]},
                            {(const float*)d_in[15], (const float*)d_in[19]}};
  const float* W_lin = (const float*)d_in[20];
  const float* b_lin = (const float*)d_in[21];
  const float* start_trans = (const float*)d_in[22];
  const float* end_trans = (const float*)d_in[23];
  const float* trans = (const float*)d_in[24];

  char* w = (char*)d_ws;
  size_t off = 0;
  auto alloc = [&](size_t bytes) -> void* {
    off = (off + 255) & ~(size_t)255;
    void* p = w + off;
    off += bytes;
    return p;
  };
  __bf16* X0 = (__bf16*)alloc((size_t)BS_ * K0_ * 2);
  __bf16* X1 = (__bf16*)alloc((size_t)BS_ * K1_ * 2);
  __bf16* X2 = (__bf16*)alloc((size_t)BS_ * K1_ * 2);
  float* pre_f = (float*)alloc((size_t)BS_ * G_ * 4);
  float* pre_b = (float*)alloc((size_t)BS_ * G_ * 4);
  float* gbuf = (float*)alloc((size_t)2 * B_ * G_ * 4);
  float* cst = (float*)alloc((size_t)2 * B_ * H_ * 4);
  __bf16* hbf = (__bf16*)alloc((size_t)2 * B_ * KH_ * 2);
  float* em = (float*)alloc((size_t)BS_ * TP_ * 4);
  __bf16* WbIh[2][2];
  __bf16* WbHh[2][2];
  float* biasLD[2][2];
  for (int l = 0; l < 2; ++l)
    for (int d = 0; d < 2; ++d) {
      int kin = (l == 0) ? K0_ : K1_;
      WbIh[l][d] = (__bf16*)alloc((size_t)kin * G_ * 2);
      WbHh[l][d] = (__bf16*)alloc((size_t)KH_ * G_ * 2);
      biasLD[l][d] = (float*)alloc((size_t)G_ * 4);
    }
  __bf16* WbLin = (__bf16*)alloc((size_t)K1_ * TP_ * 2);
  float* biasEm = (float*)alloc((size_t)TP_ * 4);
  float* llk = (float*)alloc((size_t)B_ * 4);

  const int TB = 256;
  // weight packing + bias sums
  for (int l = 0; l < 2; ++l)
    for (int d = 0; d < 2; ++d) {
      int korig = (l == 0) ? EMB_ : H2_;
      int kp = (l == 0) ? K0_ : K1_;
      pack_wT_k<<<(kp * G_ + TB - 1) / TB, TB, 0, stream>>>(Wih[l][d], WbIh[l][d],
                                                           korig, kp, G_, G_);
      pack_wT_k<<<(KH_ * G_ + TB - 1) / TB, TB, 0, stream>>>(Whh[l][d], WbHh[l][d],
                                                            H_, KH_, G_, G_);
      bias_sum_k<<<(G_ + TB - 1) / TB, TB, 0, stream>>>(bih[l][d], bhh[l][d],
                                                        biasLD[l][d], G_);
    }
  pack_wT_k<<<(K1_ * TP_ + TB - 1) / TB, TB, 0, stream>>>(W_lin, WbLin, H2_, K1_, T_, TP_);
  pad_bias_k<<<1, TP_, 0, stream>>>(b_lin, biasEm, T_, TP_);

  // embedding (writes every element incl. padding)
  embed_gather_k<<<(BS_ * K0_) / TB, TB, 0, stream>>>(x, emb, X0);
  // zero layer-output buffers (pad columns must stay zero)
  {
    long n1 = (long)BS_ * K1_ * 2 / 4;
    fill_zero_k<<<(unsigned)((n1 + TB - 1) / TB), TB, 0, stream>>>((float*)X1, n1);
    fill_zero_k<<<(unsigned)((n1 + TB - 1) / TB), TB, 0, stream>>>((float*)X2, n1);
  }

  const int gemmBlocks800 = (BS_ / 16) * (G_ / 16) / 8;  // 12800 blocks of 8 waves
  for (int l = 0; l < 2; ++l) {
    // parallel input projections for all timesteps (both directions)
    if (l == 0) {
      gemm_bf16_k<K0_><<<gemmBlocks800, TB, 0, stream>>>(X0, WbIh[0][0], nullptr, pre_f,
                                                         BS_, G_);
      gemm_bf16_k<K0_><<<gemmBlocks800, TB, 0, stream>>>(X0, WbIh[0][1], nullptr, pre_b,
                                                         BS_, G_);
    } else {
      gemm_bf16_k<K1_><<<gemmBlocks800, TB, 0, stream>>>(X1, WbIh[1][0], nullptr, pre_f,
                                                         BS_, G_);
      gemm_bf16_k<K1_><<<gemmBlocks800, TB, 0, stream>>>(X1, WbIh[1][1], nullptr, pre_b,
                                                         BS_, G_);
    }
    // reset recurrent state (h bf16 incl. K-padding, c f32)
    {
      long nh = (long)2 * B_ * KH_ * 2 / 4;
      fill_zero_k<<<(unsigned)((nh + TB - 1) / TB), TB, 0, stream>>>((float*)hbf, nh);
      long nc = (long)2 * B_ * H_;
      fill_zero_k<<<(unsigned)((nc + TB - 1) / TB), TB, 0, stream>>>(cst, nc);
    }
    __bf16* Xout = (l == 0) ? X1 : X2;
    for (int t = 0; t < S_; ++t) {
      lstm_step_gemm_k<<<dim3(G_ / 16, 2), 128, 0, stream>>>(
          hbf, WbHh[l][0], WbHh[l][1], pre_f, pre_b, biasLD[l][0], biasLD[l][1], gbuf, t);
      lstm_step_point_k<<<(2 * B_ * H_ + TB - 1) / TB, TB, 0, stream>>>(gbuf, cst, hbf,
                                                                        Xout, t);
    }
  }

  // emissions: (BS x 416) @ (416 x 32) + b_lin (padded)
  const int gemmBlocksEm = (BS_ / 16) * (TP_ / 16) / 8;  // 512
  gemm_bf16_k<K1_><<<gemmBlocksEm, TB, 0, stream>>>(X2, WbLin, biasEm, em, BS_, TP_);

  crf_k<<<B_, 32, 0, stream>>>(em, y, mask, start_trans, end_trans, trans, llk);
  final_sum_k<<<1, 32, 0, stream>>>(llk, (float*)d_out);
  (void)in_sizes; (void)n_in; (void)out_size; (void)ws_size;
}
